// TransformerBlock_51221779972425
// MI455X (gfx1250) — matrix operand order repaired and checked by reading, not compiled, not run on hardware
//
#include <hip/hip_runtime.h>

// Shapes (fixed by the reference)
#define BB 2
#define SS 2048
#define DD 1024
#define HH 16
#define DHH 64
#define ROWS (BB*SS)        // 4096
#define GELU_C 0.7978845608f

typedef unsigned short u16;
typedef unsigned int   u32;
typedef __attribute__((ext_vector_type(16))) __bf16 v16bf;
typedef __attribute__((ext_vector_type(8)))  float  v8f;

union FragU { v16bf v; uint4 q[2]; u16 s[16]; };

__device__ __forceinline__ u16 f2bf(float f) {
    u32 u = __float_as_uint(f);
    u32 r = u + 0x7fffu + ((u >> 16) & 1u);   // round-to-nearest-even
    return (u16)(r >> 16);
}

__device__ __forceinline__ v8f wmma_bf16(v16bf a, v16bf b, v8f c) {
    return __builtin_amdgcn_wmma_f32_16x16x32_bf16(false, a, false, b, (short)0, c, false, false);
}

// Load one 32-byte-per-lane fragment (two b128s).
__device__ __forceinline__ v16bf ld_frag16(const u16* p) {
    FragU f;
    f.q[0] = *(const uint4*)p;
    f.q[1] = *(const uint4*)(p + 16);
    return f.v;
}

// One wave computes a 32(M) x 64(N) tile. Ping-pong double buffering
// (K-loop unrolled by 2): fragments for step k+32/k+64 are in flight while
// the 8 wmma of step k execute; no register copies, no WAR hazard nops.
// Requires K % 64 == 0 (true for K = 1024 and 4096 here).
__device__ __forceinline__ void gemm32x64(const u16* A, const u16* Bt, int K,
                                          size_t lda, size_t ldb,
                                          int row0, int col0, int lane, v8f acc[2][4]) {
#pragma unroll
    for (int i = 0; i < 2; ++i)
#pragma unroll
        for (int t = 0; t < 4; ++t) acc[i][t] = {};

    // Per-fragment base pointers (loop-invariant).
    const u16* pa0 = A + (size_t)(row0 + (lane & 15)) * lda + ((lane >> 4) << 3);
    const u16* pa1 = pa0 + 16 * lda;
    const u16* pb[4];
#pragma unroll
    for (int t = 0; t < 4; ++t)
        pb[t] = Bt + (size_t)(col0 + 16 * t + (lane & 15)) * ldb + ((lane >> 4) << 3);

    v16bf a0[2], b0[4], a1[2], b1[4];
    a0[0] = ld_frag16(pa0);
    a0[1] = ld_frag16(pa1);
#pragma unroll
    for (int t = 0; t < 4; ++t) b0[t] = ld_frag16(pb[t]);

    int k = 0;
    for (; k + 64 < K; k += 64) {
        a1[0] = ld_frag16(pa0 + k + 32);
        a1[1] = ld_frag16(pa1 + k + 32);
#pragma unroll
        for (int t = 0; t < 4; ++t) b1[t] = ld_frag16(pb[t] + k + 32);
#pragma unroll
        for (int t = 0; t < 4; ++t) {
            acc[0][t] = wmma_bf16(a0[0], b0[t], acc[0][t]);
            acc[1][t] = wmma_bf16(a0[1], b0[t], acc[1][t]);
        }
        a0[0] = ld_frag16(pa0 + k + 64);
        a0[1] = ld_frag16(pa1 + k + 64);
#pragma unroll
        for (int t = 0; t < 4; ++t) b0[t] = ld_frag16(pb[t] + k + 64);
#pragma unroll
        for (int t = 0; t < 4; ++t) {
            acc[0][t] = wmma_bf16(a1[0], b1[t], acc[0][t]);
            acc[1][t] = wmma_bf16(a1[1], b1[t], acc[1][t]);
        }
    }
    // Tail: steps k and k+32 remain (K/32 even).
    a1[0] = ld_frag16(pa0 + k + 32);
    a1[1] = ld_frag16(pa1 + k + 32);
#pragma unroll
    for (int t = 0; t < 4; ++t) b1[t] = ld_frag16(pb[t] + k + 32);
#pragma unroll
    for (int t = 0; t < 4; ++t) {
        acc[0][t] = wmma_bf16(a0[0], b0[t], acc[0][t]);
        acc[1][t] = wmma_bf16(a0[1], b0[t], acc[1][t]);
    }
#pragma unroll
    for (int t = 0; t < 4; ++t) {
        acc[0][t] = wmma_bf16(a1[0], b1[t], acc[0][t]);
        acc[1][t] = wmma_bf16(a1[1], b1[t], acc[1][t]);
    }
}

// ------------- weight convert + transpose, LDS-tiled: W[K][N] f32 -> Wt[N][K] bf16 -------------
__global__ __launch_bounds__(256) void k_wconv(const float* __restrict__ W, u16* __restrict__ Wt,
                                               int K, int N) {
    __shared__ float tile[32][33];
    int tx = threadIdx.x & 31, ty = threadIdx.x >> 5;   // 32 x 8
    int kt = blockIdx.y * 32, nt = blockIdx.x * 32;
#pragma unroll
    for (int i = 0; i < 4; ++i)
        tile[ty * 4 + i][tx] = W[(size_t)(kt + ty * 4 + i) * N + nt + tx];
    __syncthreads();
#pragma unroll
    for (int i = 0; i < 4; ++i)
        Wt[(size_t)(nt + ty * 4 + i) * K + kt + tx] = f2bf(tile[tx][ty * 4 + i]);
}

// ---------------- layernorm (ddof=1, eps added to std) f32 -> bf16 ----------------
__global__ __launch_bounds__(256) void k_layernorm(const float* __restrict__ X,
                                                   const float* __restrict__ gamma,
                                                   const float* __restrict__ beta,
                                                   u16* __restrict__ Y) {
    __shared__ float s1[256], s2[256];
    int row = blockIdx.x, tid = threadIdx.x;
    const float* x = X + (size_t)row * DD;
    float v[4], sum = 0.f, sq = 0.f;
#pragma unroll
    for (int j = 0; j < 4; ++j) { v[j] = x[tid + 256 * j]; sum += v[j]; sq += v[j] * v[j]; }
    s1[tid] = sum; s2[tid] = sq;
    __syncthreads();
    for (int off = 128; off > 0; off >>= 1) {
        if (tid < off) { s1[tid] += s1[tid + off]; s2[tid] += s2[tid + off]; }
        __syncthreads();
    }
    float mean = s1[0] * (1.0f / DD);
    float var  = (s2[0] - (float)DD * mean * mean) * (1.0f / (DD - 1));
    float inv  = 1.0f / (sqrtf(var) + 1e-5f);
#pragma unroll
    for (int j = 0; j < 4; ++j) {
        int c = tid + 256 * j;
        Y[(size_t)row * DD + c] = f2bf(gamma[c] * ((v[j] - mean) * inv) + beta[c]);
    }
}

// ---------------- QKV GEMM + scatter (q,k: [B,H,S,DH]; v transposed: [B,H,DH,S]) ----------------
__global__ __launch_bounds__(256) void k_gemm_qkv(const u16* __restrict__ A, const u16* __restrict__ Bt,
                                                  const float* __restrict__ bias,
                                                  u16* __restrict__ q, u16* __restrict__ kk,
                                                  u16* __restrict__ vT) {
    int lane = threadIdx.x & 31, wave = threadIdx.x >> 5;
    int row0 = blockIdx.y * 256 + wave * 32;
    int col0 = blockIdx.x * 64;
    v8f acc[2][4];
    gemm32x64(A, Bt, DD, DD, DD, row0, col0, lane, acc);
    int mloc = (lane >> 4) << 3;
#pragma unroll
    for (int t = 0; t < 4; ++t) {
        int n = col0 + 16 * t + (lane & 15);
        float bv = bias[n];
        int h = n / 192, rem = n % 192, which = rem >> 6, d = rem & 63;
#pragma unroll
        for (int i = 0; i < 2; ++i)
#pragma unroll
            for (int r = 0; r < 8; ++r) {
                int row = row0 + 16 * i + mloc + r;
                int b = row >> 11, s = row & (SS - 1);
                u16 o = f2bf(acc[i][t][r] + bv);
                size_t bh = (size_t)b * HH + h;
                if (which == 0)      q [(bh * SS + s) * DHH + d] = o;
                else if (which == 1) kk[(bh * SS + s) * DHH + d] = o;
                else                 vT[(bh * DHH + d) * SS + s] = o;
            }
    }
}

// ---------------- flash attention: one 16-row q tile per wave, online softmax ----------------
__global__ __launch_bounds__(256) void k_attn(const u16* __restrict__ Q, const u16* __restrict__ K,
                                              const u16* __restrict__ VT, u16* __restrict__ O) {
    __shared__ u16 lds[8][16 * 32];
    int lane = threadIdx.x & 31, wave = threadIdx.x >> 5;
    int tile = blockIdx.x * 8 + wave;      // 0..4095
    int qt = tile & 127;                   // q tile within (b,h)
    int bh = tile >> 7;                    // b*H + h
    int qs = qt * 16;
    const u16* qh = Q  + (size_t)bh * SS * DHH;
    const u16* kh = K  + (size_t)bh * SS * DHH;
    const u16* vh = VT + (size_t)bh * DHH * SS;

    v16bf qa[2];
#pragma unroll
    for (int c = 0; c < 2; ++c)
        qa[c] = ld_frag16(qh + (size_t)(qs + (lane & 15)) * DHH + 32 * c + ((lane >> 4) << 3));

    auto load_kf = [&](int kt, v16bf out[2][2]) {
#pragma unroll
        for (int t = 0; t < 2; ++t)
#pragma unroll
            for (int c = 0; c < 2; ++c)
                out[t][c] = ld_frag16(kh + (size_t)(kt + 16 * t + (lane & 15)) * DHH
                                         + 32 * c + ((lane >> 4) << 3));
    };

    float mrow[8], lrow[8];
    v8f oacc[4];
    oacc[0] = {}; oacc[1] = {}; oacc[2] = {}; oacc[3] = {};
#pragma unroll
    for (int r = 0; r < 8; ++r) { mrow[r] = -1e30f; lrow[r] = 0.f; }
    int mbase = qs + ((lane >> 4) << 3);
    int kend = qs + 16;

    v16bf kf[2][2];
    load_kf(0, kf);

    for (int kt = 0; kt < kend; kt += 32) {
        // V fragments for this step: issued early so softmax VALU hides their latency.
        v16bf vf[4];
#pragma unroll
        for (int t = 0; t < 4; ++t)
            vf[t] = ld_frag16(vh + (size_t)(16 * t + (lane & 15)) * SS + kt + ((lane >> 4) << 3));
        // Prefetch next step's K fragments (double buffer).
        bool more = (kt + 32 < kend);
        v16bf kfn[2][2];
        if (more) load_kf(kt + 32, kfn);

        v8f sc[2]; sc[0] = {}; sc[1] = {};
#pragma unroll
        for (int t = 0; t < 2; ++t)
#pragma unroll
            for (int c = 0; c < 2; ++c)
                sc[t] = wmma_bf16(qa[c], kf[t][c], sc[t]);

        int lrbase = (lane >> 4) << 3;
#pragma unroll
        for (int r = 0; r < 8; ++r) {
            int m  = mbase + r;
            int n0 = kt + (lane & 15);
            float s0 = (n0      <= m) ? sc[0][r] * 0.125f : -1e30f;
            float s1 = (n0 + 16 <= m) ? sc[1][r] * 0.125f : -1e30f;
            float v = fmaxf(s0, s1);
            v = fmaxf(v, __shfl_xor(v, 1, 32));
            v = fmaxf(v, __shfl_xor(v, 2, 32));
            v = fmaxf(v, __shfl_xor(v, 4, 32));
            v = fmaxf(v, __shfl_xor(v, 8, 32));
            float mn = fmaxf(mrow[r], v);
            float al = __expf(mrow[r] - mn);
            mrow[r] = mn;
            float e0 = __expf(s0 - mn), e1 = __expf(s1 - mn);
            float ps = e0 + e1;
            ps += __shfl_xor(ps, 1, 32);
            ps += __shfl_xor(ps, 2, 32);
            ps += __shfl_xor(ps, 4, 32);
            ps += __shfl_xor(ps, 8, 32);
            lrow[r] = lrow[r] * al + ps;
            oacc[0][r] *= al; oacc[1][r] *= al; oacc[2][r] *= al; oacc[3][r] *= al;
            lds[wave][(lrbase + r) * 32 +      (lane & 15)] = f2bf(e0);
            lds[wave][(lrbase + r) * 32 + 16 + (lane & 15)] = f2bf(e1);
        }
        asm volatile("s_wait_dscnt 0" ::: "memory");   // wave-private LDS staging: order store->load
        FragU pf;
        const u16* lp = &lds[wave][(lane & 15) * 32 + ((lane >> 4) << 3)];
        pf.q[0] = *(const uint4*)lp;
        pf.q[1] = *(const uint4*)(lp + 16);
#pragma unroll
        for (int t = 0; t < 4; ++t)
            oacc[t] = wmma_bf16(pf.v, vf[t], oacc[t]);

        if (more) {
#pragma unroll
            for (int t = 0; t < 2; ++t)
#pragma unroll
                for (int c = 0; c < 2; ++c) kf[t][c] = kfn[t][c];
        }
    }

    int b = bh >> 4, h = bh & 15;
#pragma unroll
    for (int t = 0; t < 4; ++t) {
        int d = 16 * t + (lane & 15);
#pragma unroll
        for (int r = 0; r < 8; ++r) {
            int m = mbase + r;
            O[((size_t)b * SS + m) * DD + h * DHH + d] = f2bf(oacc[t][r] / lrow[r]);
        }
    }
}

// ---------------- O projection: + bias + residual(x) -> f32 ----------------
__global__ __launch_bounds__(256) void k_gemm_o(const u16* __restrict__ A, const u16* __restrict__ Bt,
                                                const float* __restrict__ bias, const float* __restrict__ x,
                                                float* __restrict__ attn_out) {
    int lane = threadIdx.x & 31, wave = threadIdx.x >> 5;
    int row0 = blockIdx.y * 256 + wave * 32;
    int col0 = blockIdx.x * 64;
    v8f acc[2][4];
    gemm32x64(A, Bt, DD, DD, DD, row0, col0, lane, acc);
    int mloc = (lane >> 4) << 3;
#pragma unroll
    for (int t = 0; t < 4; ++t) {
        int n = col0 + 16 * t + (lane & 15);
        float bv = bias[n];
#pragma unroll
        for (int i = 0; i < 2; ++i)
#pragma unroll
            for (int r = 0; r < 8; ++r) {
                size_t idx = (size_t)(row0 + 16 * i + mloc + r) * DD + n;
                attn_out[idx] = acc[i][t][r] + bv + x[idx];
            }
    }
}

// ---------------- MLP up: + bias, gelu -> bf16 ----------------
__global__ __launch_bounds__(256) void k_gemm_up(const u16* __restrict__ A, const u16* __restrict__ Bt,
                                                 const float* __restrict__ bias, u16* __restrict__ U) {
    int lane = threadIdx.x & 31, wave = threadIdx.x >> 5;
    int row0 = blockIdx.y * 256 + wave * 32;
    int col0 = blockIdx.x * 64;
    v8f acc[2][4];
    gemm32x64(A, Bt, DD, DD, DD, row0, col0, lane, acc);
    int mloc = (lane >> 4) << 3;
#pragma unroll
    for (int t = 0; t < 4; ++t) {
        int n = col0 + 16 * t + (lane & 15);
        float bv = bias[n];
#pragma unroll
        for (int i = 0; i < 2; ++i)
#pragma unroll
            for (int r = 0; r < 8; ++r) {
                float v = acc[i][t][r] + bv;
                float g = 0.5f * v * (1.0f + tanhf(GELU_C * (v + 0.044715f * v * v * v)));
                U[(size_t)(row0 + 16 * i + mloc + r) * (4 * DD) + n] = f2bf(g);
            }
    }
}

// ---------------- MLP down: + bias + residual(attn_out) -> f32 output ----------------
__global__ __launch_bounds__(256) void k_gemm_down(const u16* __restrict__ A, const u16* __restrict__ Bt,
                                                   const float* __restrict__ bias,
                                                   const float* __restrict__ attn_out,
                                                   float* __restrict__ out) {
    int lane = threadIdx.x & 31, wave = threadIdx.x >> 5;
    int row0 = blockIdx.y * 256 + wave * 32;
    int col0 = blockIdx.x * 64;
    v8f acc[2][4];
    gemm32x64(A, Bt, 4 * DD, 4 * DD, 4 * DD, row0, col0, lane, acc);
    int mloc = (lane >> 4) << 3;
#pragma unroll
    for (int t = 0; t < 4; ++t) {
        int n = col0 + 16 * t + (lane & 15);
        float bv = bias[n];
#pragma unroll
        for (int i = 0; i < 2; ++i)
#pragma unroll
            for (int r = 0; r < 8; ++r) {
                size_t idx = (size_t)(row0 + 16 * i + mloc + r) * DD + n;
                out[idx] = acc[i][t][r] + bv + attn_out[idx];
            }
    }
}

extern "C" void kernel_launch(void* const* d_in, const int* in_sizes, int n_in,
                              void* d_out, int out_size, void* d_ws, size_t ws_size,
                              hipStream_t stream) {
    const float* x      = (const float*)d_in[0];
    const float* gamma1 = (const float*)d_in[1];
    const float* beta1  = (const float*)d_in[2];
    const float* W_qkv  = (const float*)d_in[3];
    const float* b_qkv  = (const float*)d_in[4];
    const float* W_o    = (const float*)d_in[5];
    const float* b_o    = (const float*)d_in[6];
    const float* gamma2 = (const float*)d_in[7];
    const float* beta2  = (const float*)d_in[8];
    const float* W_up   = (const float*)d_in[9];
    const float* b_up   = (const float*)d_in[10];
    const float* W_down = (const float*)d_in[11];
    const float* b_down = (const float*)d_in[12];
    float* out = (float*)d_out;

    char* w = (char*)d_ws;
    auto alloc = [&](size_t bytes) { void* p = (void*)w; w += (bytes + 255) & ~(size_t)255; return p; };
    u16*   Wqkv_t  = (u16*)  alloc((size_t)3 * DD * DD * 2);       // [3072][1024]
    u16*   Wo_t    = (u16*)  alloc((size_t)DD * DD * 2);           // [1024][1024]
    u16*   Wup_t   = (u16*)  alloc((size_t)4 * DD * DD * 2);       // [4096][1024]
    u16*   Wdown_t = (u16*)  alloc((size_t)4 * DD * DD * 2);       // [1024][4096]
    u16*   h_bf    = (u16*)  alloc((size_t)ROWS * DD * 2);
    u16*   q_bf    = (u16*)  alloc((size_t)BB * HH * SS * DHH * 2);
    u16*   k_bf    = (u16*)  alloc((size_t)BB * HH * SS * DHH * 2);
    u16*   vT_bf   = (u16*)  alloc((size_t)BB * HH * DHH * SS * 2);
    u16*   o_bf    = (u16*)  alloc((size_t)ROWS * DD * 2);
    float* attn    = (float*)alloc((size_t)ROWS * DD * 4);
    u16*   h2_bf   = (u16*)  alloc((size_t)ROWS * DD * 2);
    u16*   u_bf    = (u16*)  alloc((size_t)ROWS * 4 * DD * 2);

    k_wconv<<<dim3(96, 32),  256, 0, stream>>>(W_qkv,  Wqkv_t,  DD,     3 * DD);
    k_wconv<<<dim3(32, 32),  256, 0, stream>>>(W_o,    Wo_t,    DD,     DD);
    k_wconv<<<dim3(128, 32), 256, 0, stream>>>(W_up,   Wup_t,   DD,     4 * DD);
    k_wconv<<<dim3(32, 128), 256, 0, stream>>>(W_down, Wdown_t, 4 * DD, DD);

    k_layernorm<<<ROWS, 256, 0, stream>>>(x, gamma1, beta1, h_bf);
    k_gemm_qkv<<<dim3(48, 16), 256, 0, stream>>>(h_bf, Wqkv_t, b_qkv, q_bf, k_bf, vT_bf);
    k_attn<<<512, 256, 0, stream>>>(q_bf, k_bf, vT_bf, o_bf);
    k_gemm_o<<<dim3(16, 16), 256, 0, stream>>>(o_bf, Wo_t, b_o, x, attn);
    k_layernorm<<<ROWS, 256, 0, stream>>>(attn, gamma2, beta2, h2_bf);
    k_gemm_up<<<dim3(64, 16), 256, 0, stream>>>(h2_bf, Wup_t, b_up, u_bf);
    k_gemm_down<<<dim3(16, 16), 256, 0, stream>>>(u_bf, Wdown_t, b_down, attn, out);
}
